// GATEncoder_86028194938978
// MI455X (gfx1250) — matrix operand order repaired and checked by reading, not compiled, not run on hardware
//
#include <hip/hip_runtime.h>
#include <cstdint>

#define NN    20000      // nodes
#define NE    320000     // edges (before self loops)
#define HEADS 4
#define CHV   64         // per-head channels
#define DIM   256        // HEADS*CHV
#define OUTD  64
#define NEG_SLOPE 0.2f
#define LN_EPS 1e-5f

typedef __attribute__((ext_vector_type(16))) _Float16 v16h;
typedef __attribute__((ext_vector_type(8)))  float    v8f;

// ---------------------------------------------------------------- utilities
__global__ void k_zero_f(float* __restrict__ p, int n) {
  int i = blockIdx.x * blockDim.x + threadIdx.x;
  if (i < n) p[i] = 0.0f;
}

__device__ __forceinline__ unsigned f2key(float f) {
  unsigned u = __float_as_uint(f);
  return (u & 0x80000000u) ? ~u : (u | 0x80000000u);
}
__device__ __forceinline__ float key2f(unsigned k) {
  unsigned u = (k & 0x80000000u) ? (k & 0x7FFFFFFFu) : ~k;
  return __uint_as_float(u);
}

// --------------------------------------------------------- fragment packing
// A fragments: for each (tile_m, kc, lane, e) store f16 of A[row][k], where
// row/k follow the CDNA5 16-bit A 16x32 per-lane layout. Zero-padded OOB.
__global__ void k_pack_a(const float* __restrict__ A, _Float16* __restrict__ F,
                         int M, int K, int kchunks, int total) {
  int i = blockIdx.x * blockDim.x + threadIdx.x;
  if (i >= total) return;
  int e    = i & 15;
  int lane = (i >> 4) & 31;
  int rest = i >> 9;
  int kc   = rest % kchunks;
  int tm   = rest / kchunks;
  int hs = lane >> 4, l15 = lane & 15;
  int v = e >> 1, w = e & 1;
  int row = tm * 16 + l15;
  int k   = kc * 32 + ((v >> 2) << 4) + (hs << 3) + ((v & 3) << 1) + w;
  F[i] = (row < M && k < K) ? (_Float16)A[row * K + k] : (_Float16)0.0f;
}

// B fragments (weights): B[k][n] = W[n][k]; per-lane layout: n = lane&15,
// k = kc*32 + (e>>3)*16 + (lane>>4)*8 + (e&7).
__global__ void k_pack_w(const float* __restrict__ W, _Float16* __restrict__ F,
                         int Nout, int K, int kchunks, int total) {
  int i = blockIdx.x * blockDim.x + threadIdx.x;
  if (i >= total) return;
  int e    = i & 15;
  int lane = (i >> 4) & 31;
  int rest = i >> 9;
  int kc   = rest % kchunks;
  int tn   = rest / kchunks;
  int col = tn * 16 + (lane & 15);
  int k   = kc * 32 + ((e >> 3) << 4) + ((lane >> 4) << 3) + (e & 7);
  F[i] = (col < Nout && k < K) ? (_Float16)W[col * K + k] : (_Float16)0.0f;
}

// ---------------------------------------------------- WMMA GEMM (fragments)
// Out[M,Nout] = A @ W^T from pre-packed f16 fragments. One wave computes a
// 16x64 strip: 4 N-tiles share one A fragment -> 4 WMMAs per K-chunk.
// Software-pipelined: fragment loads for chunk kc+1 are issued before the
// WMMAs of chunk kc (tail index clamped, branch-free), so vmem latency
// overlaps the matrix pipe.
__global__ __launch_bounds__(256)
void k_gemm_wmma(const _Float16* __restrict__ Af, const _Float16* __restrict__ Wf,
                 float* __restrict__ Out, int M, int Nout, int kchunks) {
  const int lane    = threadIdx.x & 31;
  const int wave    = threadIdx.x >> 5;
  const int tiles_n = Nout >> 4;           // 16
  const int strips  = tiles_n >> 2;        // 4 strips of 4 tiles
  const int tiles_m = (M + 15) >> 4;
  const int wid     = blockIdx.x * 8 + wave;
  if (wid >= tiles_m * strips) return;     // wave-uniform exit
  const int tm  = wid / strips;
  const int tn0 = (wid - tm * strips) * 4;
  const int hs  = lane >> 4;
  const int l15 = lane & 15;

  // fragment pointers; element stride between k-chunks = 32 v16h (one/lane)
  const v16h* ap = (const v16h*)(Af + (size_t)(tm * kchunks) * 512) + lane;
  const v16h* b0 = (const v16h*)(Wf + (size_t)((tn0 + 0) * kchunks) * 512) + lane;
  const v16h* b1 = (const v16h*)(Wf + (size_t)((tn0 + 1) * kchunks) * 512) + lane;
  const v16h* b2 = (const v16h*)(Wf + (size_t)((tn0 + 2) * kchunks) * 512) + lane;
  const v16h* b3 = (const v16h*)(Wf + (size_t)((tn0 + 3) * kchunks) * 512) + lane;

  v8f acc0 = {}, acc1 = {}, acc2 = {}, acc3 = {};
  v16h a = ap[0], w0 = b0[0], w1 = b1[0], w2 = b2[0], w3 = b3[0];

  for (int kc = 0; kc < kchunks; ++kc) {
    const int kn = (kc + 1 < kchunks) ? (kc + 1) : kc;   // clamped (branch-free)
    const int kp = (kc + 2 < kchunks) ? (kc + 2) : kc;
    __builtin_prefetch((const void*)(ap + kp * 32), 0, 1);
    // issue next chunk's loads before this chunk's WMMAs
    v16h an  = ap[kn * 32];
    v16h w0n = b0[kn * 32];
    v16h w1n = b1[kn * 32];
    v16h w2n = b2[kn * 32];
    v16h w3n = b3[kn * 32];
    acc0 = __builtin_amdgcn_wmma_f32_16x16x32_f16(false, a, false, w0, (short)0, acc0, false, false);
    acc1 = __builtin_amdgcn_wmma_f32_16x16x32_f16(false, a, false, w1, (short)0, acc1, false, false);
    acc2 = __builtin_amdgcn_wmma_f32_16x16x32_f16(false, a, false, w2, (short)0, acc2, false, false);
    acc3 = __builtin_amdgcn_wmma_f32_16x16x32_f16(false, a, false, w3, (short)0, acc3, false, false);
    a = an; w0 = w0n; w1 = w1n; w2 = w2n; w3 = w3n;
  }

#pragma unroll
  for (int v = 0; v < 8; ++v) {            // C/D f32 16x16 layout
    int r = tm * 16 + v + (hs << 3);
    if (r < M) {
      float* o = Out + (size_t)r * Nout + l15;
      o[(tn0 + 0) * 16] = acc0[v];
      o[(tn0 + 1) * 16] = acc1[v];
      o[(tn0 + 2) * 16] = acc2[v];
      o[(tn0 + 3) * 16] = acc3[v];
    }
  }
}

// ------------------------------------------- self-loop mean edge attr (once)
__global__ void k_deg_sum(const long long* __restrict__ ei,
                          const float* __restrict__ eattr,
                          float* __restrict__ deg, float* __restrict__ sattr) {
  int e = blockIdx.x * blockDim.x + threadIdx.x;
  if (e >= NE) return;
  int d = (int)ei[NE + e];
  atomicAdd(&deg[d], 1.0f);
  atomicAdd(&sattr[d], eattr[e]);
}
__global__ void k_mean_attr(const float* __restrict__ deg, float* __restrict__ sattr) {
  int i = blockIdx.x * blockDim.x + threadIdx.x;
  if (i < NN) sattr[i] = sattr[i] / fmaxf(deg[i], 1.0f);
}

// ------------------------------------------------ per-node attention scalars
__global__ void k_node_att(const float* __restrict__ xl,
                           const float* __restrict__ att_src,
                           const float* __restrict__ att_dst,
                           float* __restrict__ a_src, float* __restrict__ a_dst) {
  int i = blockIdx.x * blockDim.x + threadIdx.x;
  if (i >= NN * HEADS) return;
  int node = i >> 2, h = i & 3;
  const float* row = xl + node * DIM + h * CHV;
  const float* as = att_src + h * CHV;
  const float* ad = att_dst + h * CHV;
  float s = 0.0f, d = 0.0f;
  for (int c = 0; c < CHV; ++c) { float v = row[c]; s += v * as[c]; d += v * ad[c]; }
  a_src[i] = s; a_dst[i] = d;
}

// a_edge[e][h] = eattr[e] * S[h],  S[h] = dot(lin_edge_w[h*64:], att_edge[h*64:])
__global__ void k_edge_scale(const float* __restrict__ lew,
                             const float* __restrict__ ae, float* __restrict__ S) {
  int h = threadIdx.x;
  if (h >= HEADS) return;
  float s = 0.0f;
  for (int c = 0; c < CHV; ++c) s += lew[h * CHV + c] * ae[h * CHV + c];
  S[h] = s;
}

// ----------------------------------------- edge passes (E real + NN self loops)
__global__ void k_edge_alpha(const long long* __restrict__ ei,
                             const float* __restrict__ eattr,
                             const float* __restrict__ mattr,
                             const float* __restrict__ a_src,
                             const float* __restrict__ a_dst,
                             const float* __restrict__ S,
                             float* __restrict__ alpha, unsigned* __restrict__ amax) {
  int i = blockIdx.x * blockDim.x + threadIdx.x;
  if (i >= (NE + NN) * HEADS) return;
  int e = i >> 2, h = i & 3;
  int s, d; float ea;
  if (e < NE) { s = (int)ei[e]; d = (int)ei[NE + e]; ea = eattr[e]; }
  else        { s = e - NE; d = s; ea = mattr[s]; }
  float a = a_src[s * HEADS + h] + a_dst[d * HEADS + h] + ea * S[h];
  a = (a > 0.0f) ? a : NEG_SLOPE * a;              // leaky_relu
  alpha[i] = a;
  atomicMax(&amax[d * HEADS + h], f2key(a));       // segment max (ordered keys)
}

__global__ void k_edge_exp(const long long* __restrict__ ei,
                           float* __restrict__ alpha,
                           const unsigned* __restrict__ amax,
                           float* __restrict__ denom) {
  int i = blockIdx.x * blockDim.x + threadIdx.x;
  if (i >= (NE + NN) * HEADS) return;
  int e = i >> 2, h = i & 3;
  int d = (e < NE) ? (int)ei[NE + e] : (e - NE);
  float ex = __expf(alpha[i] - key2f(amax[d * HEADS + h]));
  alpha[i] = ex;                                   // reuse buffer for exp
  atomicAdd(&denom[d * HEADS + h], ex);
}

// one block per edge, 256 channels; agg[dst] += xl[src] * attn
__global__ __launch_bounds__(256)
void k_edge_agg(const long long* __restrict__ ei, const float* __restrict__ xl,
                const float* __restrict__ ex, const float* __restrict__ denom,
                float* __restrict__ agg) {
  int e = blockIdx.x, t = threadIdx.x, h = t >> 6;
  int s, d;
  if (e < NE) { s = (int)ei[e]; d = (int)ei[NE + e]; }
  else        { s = e - NE; d = s; }
  float w = ex[e * HEADS + h] / (denom[d * HEADS + h] + 1e-16f);
  atomicAdd(&agg[d * DIM + t], xl[s * DIM + t] * w);
}

// ------------------------------------ epilogues: bias + LN (+res) + ELU
__global__ __launch_bounds__(256)
void k_post_concat(const float* __restrict__ agg, const float* __restrict__ bias,
                   const float* __restrict__ g, const float* __restrict__ b,
                   float* __restrict__ h_io, int residual) {
  int wave = threadIdx.x >> 5, lane = threadIdx.x & 31;
  int node = blockIdx.x * 8 + wave;
  if (node >= NN) return;
  float v[8]; float mu = 0.0f;
#pragma unroll
  for (int j = 0; j < 8; ++j) { int c = lane + j * 32; v[j] = agg[node * DIM + c] + bias[c]; mu += v[j]; }
  for (int o = 16; o >= 1; o >>= 1) mu += __shfl_xor(mu, o, 32);
  mu *= (1.0f / DIM);
  float var = 0.0f;
#pragma unroll
  for (int j = 0; j < 8; ++j) { float t = v[j] - mu; var += t * t; }
  for (int o = 16; o >= 1; o >>= 1) var += __shfl_xor(var, o, 32);
  float rs = rsqrtf(var * (1.0f / DIM) + LN_EPS);
#pragma unroll
  for (int j = 0; j < 8; ++j) {
    int c = lane + j * 32;
    float o = (v[j] - mu) * rs * g[c] + b[c];
    if (residual) o += h_io[node * DIM + c];
    o = (o > 0.0f) ? o : (__expf(o) - 1.0f);       // elu
    h_io[node * DIM + c] = o;
  }
}

__global__ __launch_bounds__(256)
void k_post_final(const float* __restrict__ agg, const float* __restrict__ bias,
                  const float* __restrict__ g, const float* __restrict__ b,
                  float* __restrict__ out) {
  int wave = threadIdx.x >> 5, lane = threadIdx.x & 31;
  int node = blockIdx.x * 8 + wave;
  if (node >= NN) return;
  float v[2]; float mu = 0.0f;
#pragma unroll
  for (int j = 0; j < 2; ++j) {
    int c = lane + j * 32; float s = 0.0f;
#pragma unroll
    for (int h = 0; h < HEADS; ++h) s += agg[node * DIM + h * CHV + c];
    v[j] = s * 0.25f + bias[c]; mu += v[j];        // head mean
  }
  for (int o = 16; o >= 1; o >>= 1) mu += __shfl_xor(mu, o, 32);
  mu *= (1.0f / OUTD);
  float var = 0.0f;
#pragma unroll
  for (int j = 0; j < 2; ++j) { float t = v[j] - mu; var += t * t; }
  for (int o = 16; o >= 1; o >>= 1) var += __shfl_xor(var, o, 32);
  float rs = rsqrtf(var * (1.0f / OUTD) + LN_EPS);
#pragma unroll
  for (int j = 0; j < 2; ++j) {
    int c = lane + j * 32;
    float o = (v[j] - mu) * rs * g[c] + b[c];
    o = (o > 0.0f) ? o : (__expf(o) - 1.0f);
    out[node * OUTD + c] = o;
  }
}

// ---------------------------------------------------------------- launcher
extern "C" void kernel_launch(void* const* d_in, const int* in_sizes, int n_in,
                              void* d_out, int out_size, void* d_ws, size_t ws_size,
                              hipStream_t stream) {
  const float*     x     = (const float*)d_in[0];
  const float*     eattr = (const float*)d_in[1];
  const long long* ei    = (const long long*)d_in[26];   // int64 edge_index [2,E]

  struct Layer { const float *lin_w, *att_src, *att_dst, *lin_edge_w, *att_edge, *bias, *ln_g, *ln_b; };
  Layer L[3];
  for (int li = 0; li < 3; ++li) {
    int base = 2 + li * 8;
    L[li].lin_w      = (const float*)d_in[base + 0];
    L[li].att_src    = (const float*)d_in[base + 1];
    L[li].att_dst    = (const float*)d_in[base + 2];
    L[li].lin_edge_w = (const float*)d_in[base + 3];
    L[li].att_edge   = (const float*)d_in[base + 4];
    L[li].bias       = (const float*)d_in[base + 5];
    L[li].ln_g       = (const float*)d_in[base + 6];
    L[li].ln_b       = (const float*)d_in[base + 7];
  }

  // ---- workspace bump allocator (all accumulated buffers are re-zeroed)
  char* w = (char*)d_ws;
  auto alloc = [&](size_t bytes) -> void* {
    void* r = (void*)w; w += (bytes + 255) & ~(size_t)255; return r;
  };
  float*     buf_h   = (float*)alloc((size_t)NN * DIM * 4);
  float*     buf_xl  = (float*)alloc((size_t)NN * DIM * 4);
  float*     buf_agg = (float*)alloc((size_t)NN * DIM * 4);
  float*     alpha   = (float*)alloc((size_t)(NE + NN) * HEADS * 4);
  float*     a_src   = (float*)alloc((size_t)NN * HEADS * 4);
  float*     a_dst   = (float*)alloc((size_t)NN * HEADS * 4);
  unsigned*  amax    = (unsigned*)alloc((size_t)NN * HEADS * 4);
  float*     denom   = (float*)alloc((size_t)NN * HEADS * 4);
  float*     deg     = (float*)alloc((size_t)NN * 4);
  float*     mattr   = (float*)alloc((size_t)NN * 4);
  float*     S       = (float*)alloc(HEADS * 4);
  const int  tiles_m = NN / 16;                      // 1250 (exact)
  const int  tiles_n = DIM / 16;                     // 16
  const int  kch_max = DIM / 32;                     // 8
  _Float16*  a_frag  = (_Float16*)alloc((size_t)tiles_m * kch_max * 512 * 2);
  _Float16*  w_frag  = (_Float16*)alloc((size_t)tiles_n * kch_max * 512 * 2);

  const int EA = NE + NN;

  // ---- self-loop fill_value='mean' precompute
  k_zero_f<<<(NN + 255) / 256, 256, 0, stream>>>(deg, NN);
  k_zero_f<<<(NN + 255) / 256, 256, 0, stream>>>(mattr, NN);
  k_deg_sum<<<(NE + 255) / 256, 256, 0, stream>>>(ei, eattr, deg, mattr);
  k_mean_attr<<<(NN + 255) / 256, 256, 0, stream>>>(deg, mattr);

  for (int li = 0; li < 3; ++li) {
    const float* hin = (li == 0) ? x : buf_h;
    const int K       = (li == 0) ? 5 : DIM;
    const int kchunks = (K + 31) / 32;

    k_zero_f<<<(NN * DIM + 255) / 256, 256, 0, stream>>>(buf_agg, NN * DIM);
    k_zero_f<<<(NN * HEADS + 255) / 256, 256, 0, stream>>>(denom, NN * HEADS);
    k_zero_f<<<(NN * HEADS + 255) / 256, 256, 0, stream>>>((float*)amax, NN * HEADS);

    // pack A and W into WMMA fragment order (f16, zero-padded past K)
    const int a_tot = tiles_m * kchunks * 512;
    const int w_tot = tiles_n * kchunks * 512;
    k_pack_a<<<(a_tot + 255) / 256, 256, 0, stream>>>(hin, a_frag, NN, K, kchunks, a_tot);
    k_pack_w<<<(w_tot + 255) / 256, 256, 0, stream>>>(L[li].lin_w, w_frag, DIM, K, kchunks, w_tot);

    // 16x64 strip per wave: 1250 * 4 waves, 8 waves/block
    const int waves = tiles_m * (tiles_n / 4);
    k_gemm_wmma<<<(waves + 7) / 8, 256, 0, stream>>>(a_frag, w_frag, buf_xl, NN, DIM, kchunks);

    k_node_att<<<(NN * HEADS + 255) / 256, 256, 0, stream>>>(buf_xl, L[li].att_src, L[li].att_dst, a_src, a_dst);
    k_edge_scale<<<1, 32, 0, stream>>>(L[li].lin_edge_w, L[li].att_edge, S);

    k_edge_alpha<<<(EA * HEADS + 255) / 256, 256, 0, stream>>>(ei, eattr, mattr, a_src, a_dst, S, alpha, amax);
    k_edge_exp<<<(EA * HEADS + 255) / 256, 256, 0, stream>>>(ei, alpha, amax, denom);
    k_edge_agg<<<EA, 256, 0, stream>>>(ei, buf_xl, alpha, denom, buf_agg);

    if (li < 2)
      k_post_concat<<<(NN + 7) / 8, 256, 0, stream>>>(buf_agg, L[li].bias, L[li].ln_g, L[li].ln_b, buf_h, li > 0 ? 1 : 0);
    else
      k_post_final<<<(NN + 7) / 8, 256, 0, stream>>>(buf_agg, L[li].bias, L[li].ln_g, L[li].ln_b, (float*)d_out);
  }
}
